// RNNSeq2Seq_2413771621100
// MI455X (gfx1250) — matrix operand order repaired and checked by reading, not compiled, not run on hardware
//
#include <hip/hip_runtime.h>
#include <math.h>

// ---------------------------------------------------------------------------
// RNN seq2seq with Bahdanau attention on MI455X (gfx1250, wave32, WMMA).
//  * All GEMMs: v_wmma_f32_16x16x32_bf16 (A bf16, B = W[N,K] row-major, C f32).
//  * Sequential recurrences: persistent single-workgroup kernels, hidden state
//    in LDS as bf16 A-fragments (M padded 8->16), native v_tanh_f32.
//  * pred = h2 @ out_W^T hoisted out of the decode loop into one batched
//    [2048,512]x[512,32000] GEMM with 64-row M blocking (B read ~1 GB from L2,
//    out_W bf16 32 MB L2-resident, 262 MB f32 stores -> HBM-store bound) and
//    async global->LDS staging of A tiles (ASYNCcnt path).
//  * Algebra: dec layer2 uses (Wi2+Wh2); dec layer1 uses [x_t|h] @ [Wi1|Wh1]^T.
// ---------------------------------------------------------------------------

typedef __bf16 bf16_t;
typedef __attribute__((ext_vector_type(16))) bf16_t v16bf;
typedef __attribute__((ext_vector_type(8)))  float  v8f;

constexpr int BATCH = 8, SENC = 128, SDEC = 256, EDIM = 256, HDIM = 512, NVOC = 32000;
constexpr int KCAT = EDIM + HDIM + HDIM;   // 1280 (emb | ctx | h)
constexpr int A1LD = KCAT + 8;             // 1288, LDS stride (bank-conflict pad)
constexpr int HLD  = HDIM + 8;             // 520

union Frag { v16bf v; uint4 q[2]; };

__device__ inline v8f zero_v8f() {
  v8f z = {0.f, 0.f, 0.f, 0.f, 0.f, 0.f, 0.f, 0.f};
  return z;
}

// CDNA5 has native V_TANH_F32 (trans op); fall back to libm if builtin absent.
__device__ inline float fast_tanh(float x) {
#if __has_builtin(__builtin_amdgcn_tanhf)
  return __builtin_amdgcn_tanhf(x);
#else
  return tanhf(x);
#endif
}

// A fragment (16x32 bf16, M x K): lane holds row M=lane%16; group g=lane/16
// holds K chunks [kbase+g*8 .. +7] and [kbase+16+g*8 .. +7].
__device__ inline v16bf load_fragA(const bf16_t* base, int ldk, int kbase) {
  const int lane = threadIdx.x & 31;
  const int row = lane & 15, g = lane >> 4;
  const bf16_t* p = base + row * ldk + kbase + g * 8;
  Frag f;
  f.q[0] = *reinterpret_cast<const uint4*>(p);
  f.q[1] = *reinterpret_cast<const uint4*>(p + 16);
  return f.v;
}

// B fragment (32x16 bf16, K x N), B = W^T with W [N,K] row-major:
// lane holds column N=lane%16; group g holds K chunks [kbase+g*8 .. +7] and [kbase+16+g*8 .. +7].
__device__ inline v16bf load_fragB(const bf16_t* base, int ldk, int kbase) {
  const int lane = threadIdx.x & 31;
  const int col = lane & 15, g = lane >> 4;
  const bf16_t* p = base + col * ldk + kbase + g * 8;
  Frag f;
  f.q[0] = *reinterpret_cast<const uint4*>(p);
  f.q[1] = *reinterpret_cast<const uint4*>(p + 16);
  return f.v;
}

__device__ inline v8f wmma_bf16(v16bf a, v16bf b, v8f c) {
  return __builtin_amdgcn_wmma_f32_16x16x32_bf16(false, a, false, b, (short)0, c,
                                                 false, false);
}

// ------------------------------ prep kernels -------------------------------

__global__ void k_cvt_bf16(const float* __restrict__ s, bf16_t* __restrict__ d, int n) {
  int i = blockIdx.x * 256 + threadIdx.x;
  if (i < n) d[i] = (bf16_t)s[i];
}

__global__ void k_prep_dec(const float* __restrict__ Wi1, const float* __restrict__ Wh1,
                           const float* __restrict__ Wi2, const float* __restrict__ Wh2,
                           const float* __restrict__ bi1, const float* __restrict__ bh1,
                           const float* __restrict__ bi2, const float* __restrict__ bh2,
                           bf16_t* __restrict__ Wcat1, bf16_t* __restrict__ W2sum,
                           float* __restrict__ bcat1, float* __restrict__ b2sum) {
  int i = blockIdx.x * 256 + threadIdx.x;
  if (i < HDIM * KCAT) {
    int n = i / KCAT, k = i % KCAT;
    float v = (k < EDIM + HDIM) ? Wi1[n * (EDIM + HDIM) + k]
                                : Wh1[n * HDIM + (k - (EDIM + HDIM))];
    Wcat1[i] = (bf16_t)v;
  }
  if (i < HDIM * HDIM) W2sum[i] = (bf16_t)(Wi2[i] + Wh2[i]);
  if (i < HDIM) { bcat1[i] = bi1[i] + bh1[i]; b2sum[i] = bi2[i] + bh2[i]; }
}

__global__ void k_gather_emb(const int* __restrict__ tok, const float* __restrict__ emb,
                             bf16_t* __restrict__ out, int rows, int cols) {
  int i = blockIdx.x * 256 + threadIdx.x;
  if (i < rows * cols) {
    int r = i / cols, c = i % cols;
    out[i] = (bf16_t)emb[(size_t)tok[r] * cols + c];
  }
}

// ------------------ generic GEMM: C[M,N] = A[M,K] @ W[N,K]^T + bias --------
// grid: (N/256, M/16), block 256 (8 waves, 2 N-tiles each). K <= 512.

__global__ __launch_bounds__(256)
void k_gemm_bias(const bf16_t* __restrict__ A, const bf16_t* __restrict__ W,
                 const float* __restrict__ bias, float* __restrict__ C,
                 int N, int K) {
  __shared__ __align__(16) bf16_t As[16 * (512 + 8)];
  const int tid = threadIdx.x;
  const int m0 = blockIdx.y * 16;
  const int lda = K + 8;
  const int cpk = K / 8;
  for (int c = tid; c < 16 * cpk; c += 256) {
    int r = c / cpk, cc = c % cpk;
    *reinterpret_cast<uint4*>(&As[r * lda + cc * 8]) =
        *reinterpret_cast<const uint4*>(&A[(size_t)(m0 + r) * K + cc * 8]);
  }
  __syncthreads();
  const int w = tid >> 5, lane = tid & 31;
  const int n0 = blockIdx.x * 256 + w * 32;
  v8f acc0 = zero_v8f(), acc1 = zero_v8f();
  for (int kt = 0; kt < K; kt += 32) {
    v16bf a  = load_fragA(As, lda, kt);
    v16bf b0 = load_fragB(W + (size_t)n0 * K, K, kt);
    v16bf b1 = load_fragB(W + (size_t)(n0 + 16) * K, K, kt);
    acc0 = wmma_bf16(a, b0, acc0);
    acc1 = wmma_bf16(a, b1, acc1);
  }
  const int mrow = (lane >> 4) * 8, nc = lane & 15;
  for (int r = 0; r < 8; ++r) {
    int m = m0 + mrow + r;
    int n = n0 + nc;
    C[(size_t)m * N + n] = acc0[r] + bias[n];
    C[(size_t)m * N + n + 16] = acc1[r] + bias[n + 16];
  }
}

// -------- persistent Elman scan: h_t = tanh(xi[:,t,:] + h @ Wh^T + bh) -----
// single workgroup, 8 waves; state (batch padded to 16 rows) lives in LDS bf16.

__global__ __launch_bounds__(256)
void k_rnn_scan(const float* __restrict__ xi,     // [8*S][H], row = b*S+s
                const bf16_t* __restrict__ Wh,    // [H][H]
                const float* __restrict__ bh,
                const float* __restrict__ h0,     // [8][H] f32 or nullptr
                bf16_t* __restrict__ out_bf,      // [8*S][H]
                float* __restrict__ hT,           // [8][H]
                int S) {
  __shared__ __align__(16) bf16_t hbf[16 * HLD];
  const int tid = threadIdx.x;
  for (int i = tid; i < 16 * HLD; i += 256) {
    int m = i / HLD, n = i % HLD;
    float v = (h0 != nullptr && m < BATCH && n < HDIM) ? h0[m * HDIM + n] : 0.f;
    hbf[i] = (bf16_t)v;
  }
  __syncthreads();
  const int w = tid >> 5, lane = tid & 31;
  const int mrow = (lane >> 4) * 8, nc = lane & 15;
  for (int t = 0; t < S; ++t) {
    v8f acc[4];
    for (int j = 0; j < 4; ++j) acc[j] = zero_v8f();
    for (int kt = 0; kt < HDIM; kt += 32) {
      v16bf a = load_fragA(hbf, HLD, kt);
#pragma unroll
      for (int j = 0; j < 4; ++j) {
        int n0 = (w * 4 + j) * 16;
        v16bf b = load_fragB(Wh + (size_t)n0 * HDIM, HDIM, kt);
        acc[j] = wmma_bf16(a, b, acc[j]);
      }
    }
    float vals[4][8];
    for (int j = 0; j < 4; ++j) {
      int n = (w * 4 + j) * 16 + nc;
      for (int r = 0; r < 8; ++r) {
        int m = mrow + r;
        float x = acc[j][r] + bh[n];
        if (m < BATCH) x += xi[((size_t)m * S + t) * HDIM + n];
        vals[j][r] = fast_tanh(x);
      }
    }
    __syncthreads();  // all A-frag reads done before state update
    for (int j = 0; j < 4; ++j) {
      int n = (w * 4 + j) * 16 + nc;
      for (int r = 0; r < 8; ++r) {
        int m = mrow + r;
        bf16_t hv = (bf16_t)vals[j][r];
        hbf[m * HLD + n] = hv;
        if (m < BATCH) out_bf[((size_t)m * S + t) * HDIM + n] = hv;
      }
    }
    __syncthreads();
  }
  for (int i = tid; i < BATCH * HDIM; i += 256) {
    int m = i >> 9, n = i & (HDIM - 1);
    hT[i] = (float)hbf[m * HLD + n];
  }
}

// ------------------- persistent decoder (256 steps, 16 waves) --------------

__global__ __launch_bounds__(512)
void k_decoder(const int* __restrict__ target,        // [8][256]
               const float* __restrict__ dec_emb,     // [VD][E]
               const float* __restrict__ pre_enc,     // [8*128][H] f32
               const bf16_t* __restrict__ enc_out,    // [8*128][H]
               const bf16_t* __restrict__ attnWh,     // [H][H]
               const float* __restrict__ attn_bh,
               const float* __restrict__ attnV,       // [H]
               const float* __restrict__ attn_bV,     // [1]
               const bf16_t* __restrict__ Wcat1,      // [H][1280]
               const float* __restrict__ bcat1,
               const bf16_t* __restrict__ W2sum,      // [H][H]
               const float* __restrict__ b2sum,
               const float* __restrict__ h_init,      // [8][H] f32
               bf16_t* __restrict__ H2) {             // [2048][H], row = t*8+b
  extern __shared__ __align__(16) char smem[];
  bf16_t* A1   = reinterpret_cast<bf16_t*>(smem);          // [16][A1LD]
  bf16_t* h1bf = A1 + 16 * A1LD;                           // [16][HLD]
  bf16_t* hwbf = h1bf + 16 * HLD;                          // [8][HDIM]
  float*  dots = reinterpret_cast<float*>(hwbf + 8 * HDIM);// [8][128]

  const int tid = threadIdx.x;
  const int w = tid >> 5, lane = tid & 31;
  const int mrow = (lane >> 4) * 8, nc = lane & 15;

  for (int i = tid; i < 16 * A1LD; i += 512) A1[i] = (bf16_t)0.f;
  __syncthreads();
  for (int i = tid; i < BATCH * HDIM; i += 512) {
    int b = i >> 9, k = i & (HDIM - 1);
    A1[b * A1LD + EDIM + HDIM + k] = (bf16_t)h_init[i];
  }
  __syncthreads();

  for (int t = 0; t < SDEC; ++t) {
    // (a) hw = h @ attnWh^T + attn_bh   (h lives in A1 cols 768..1279)
    {
      const int n0 = w * 32;
      v8f a0 = zero_v8f(), a1 = zero_v8f();
      for (int kt = 0; kt < HDIM; kt += 32) {
        v16bf a  = load_fragA(A1, A1LD, EDIM + HDIM + kt);
        v16bf b0 = load_fragB(attnWh + (size_t)n0 * HDIM, HDIM, kt);
        v16bf b1 = load_fragB(attnWh + (size_t)(n0 + 16) * HDIM, HDIM, kt);
        a0 = wmma_bf16(a, b0, a0);
        a1 = wmma_bf16(a, b1, a1);
      }
      for (int r = 0; r < 8; ++r) {
        int m = mrow + r;
        if (m < BATCH) {
          int n = n0 + nc;
          hwbf[m * HDIM + n]      = (bf16_t)(a0[r] + attn_bh[n]);
          hwbf[m * HDIM + n + 16] = (bf16_t)(a1[r] + attn_bh[n + 16]);
        }
      }
    }
    __syncthreads();
    // (b) dots[b,s] = sum_h tanh(pre_enc + hw) * V + bV   (2 pairs/thread)
    for (int pp = tid * 2; pp < tid * 2 + 2; ++pp) {
      int b = pp >> 7, s = pp & 127;
      const float* pe = pre_enc + ((size_t)(b * SENC + s)) * HDIM;
      const bf16_t* hw = hwbf + b * HDIM;
      float acc = 0.f;
      for (int k = 0; k < HDIM; ++k)
        acc += fast_tanh(pe[k] + (float)hw[k]) * attnV[k];
      dots[pp] = acc + attn_bV[0];
    }
    __syncthreads();
    // (c) softmax over s per batch row: waves 0..7, one row each
    if (w < BATCH) {
      float* row = dots + w * SENC;
      float vv[4], mx = -1e30f;
      for (int i = 0; i < 4; ++i) { vv[i] = row[lane + 32 * i]; mx = fmaxf(mx, vv[i]); }
      for (int off = 16; off > 0; off >>= 1) mx = fmaxf(mx, __shfl_xor(mx, off, 32));
      float sum = 0.f;
      for (int i = 0; i < 4; ++i) { vv[i] = __expf(vv[i] - mx); sum += vv[i]; }
      for (int off = 16; off > 0; off >>= 1) sum += __shfl_xor(sum, off, 32);
      float inv = 1.f / sum;
      for (int i = 0; i < 4; ++i) row[lane + 32 * i] = vv[i] * inv;
    }
    __syncthreads();
    // (d) ctx -> A1 cols 256..767 ; emb -> A1 cols 0..255
    for (int o = tid; o < BATCH * HDIM; o += 512) {
      int b = o >> 9, hh = o & (HDIM - 1);
      const float* aw = dots + b * SENC;
      const bf16_t* eo = enc_out + ((size_t)b * SENC) * HDIM + hh;
      float acc = 0.f;
      for (int s = 0; s < SENC; ++s) acc += aw[s] * (float)eo[(size_t)s * HDIM];
      A1[b * A1LD + EDIM + hh] = (bf16_t)acc;
    }
    for (int o = tid; o < BATCH * EDIM; o += 512) {
      int b = o >> 8, e = o & (EDIM - 1);
      int tok = (t == 0) ? target[b * SDEC] : target[b * SDEC + t - 1];
      A1[b * A1LD + e] = (bf16_t)dec_emb[(size_t)tok * EDIM + e];
    }
    __syncthreads();
    // (e) h1 = tanh([emb|ctx|h] @ Wcat1^T + (bi1+bh1)), K = 1280
    {
      const int n0 = w * 32;
      v8f a0 = zero_v8f(), a1 = zero_v8f();
      for (int kt = 0; kt < KCAT; kt += 32) {
        v16bf a  = load_fragA(A1, A1LD, kt);
        v16bf b0 = load_fragB(Wcat1 + (size_t)n0 * KCAT, KCAT, kt);
        v16bf b1 = load_fragB(Wcat1 + (size_t)(n0 + 16) * KCAT, KCAT, kt);
        a0 = wmma_bf16(a, b0, a0);
        a1 = wmma_bf16(a, b1, a1);
      }
      for (int r = 0; r < 8; ++r) {
        int m = mrow + r;
        int n = n0 + nc;
        h1bf[m * HLD + n]      = (bf16_t)fast_tanh(a0[r] + bcat1[n]);
        h1bf[m * HLD + n + 16] = (bf16_t)fast_tanh(a1[r] + bcat1[n + 16]);
      }
    }
    __syncthreads();
    // (f) h2 = tanh(h1 @ (Wi2+Wh2)^T + (bi2+bh2)) -> next h slot + H2
    {
      const int n0 = w * 32;
      v8f a0 = zero_v8f(), a1 = zero_v8f();
      for (int kt = 0; kt < HDIM; kt += 32) {
        v16bf a  = load_fragA(h1bf, HLD, kt);
        v16bf b0 = load_fragB(W2sum + (size_t)n0 * HDIM, HDIM, kt);
        v16bf b1 = load_fragB(W2sum + (size_t)(n0 + 16) * HDIM, HDIM, kt);
        a0 = wmma_bf16(a, b0, a0);
        a1 = wmma_bf16(a, b1, a1);
      }
      for (int r = 0; r < 8; ++r) {
        int m = mrow + r;
        int n = n0 + nc;
        float v0 = fast_tanh(a0[r] + b2sum[n]);
        float v1 = fast_tanh(a1[r] + b2sum[n + 16]);
        A1[m * A1LD + EDIM + HDIM + n]      = (bf16_t)v0;
        A1[m * A1LD + EDIM + HDIM + n + 16] = (bf16_t)v1;
        if (m < BATCH) {
          H2[((size_t)t * BATCH + m) * HDIM + n]      = (bf16_t)v0;
          H2[((size_t)t * BATCH + m) * HDIM + n + 16] = (bf16_t)v1;
        }
      }
    }
    __syncthreads();
  }
}

// --------- output projection: out[b,t,:] = H2[t*8+b,:] @ outW^T + out_b ----
// grid: (NVOC/256, 2048/64); 64-row M blocking for 4x B(L2) reuse; A tiles
// staged LDS via async global->LDS loads (ASYNCcnt). HBM-store-bound.

__global__ __launch_bounds__(256)
void k_out_gemm(const bf16_t* __restrict__ H2, const bf16_t* __restrict__ Wo,
                const float* __restrict__ bo, float* __restrict__ out) {
  extern __shared__ __align__(16) char smem[];
  bf16_t* As = reinterpret_cast<bf16_t*>(smem);  // [64][HLD]
  const int tid = threadIdx.x;
  const int m0 = blockIdx.y * 64;
  // Stage A rows m0..m0+63 into LDS with async copies (no VGPR round trip).
  for (int c = tid; c < 64 * 64; c += 256) {
    int r = c >> 6, cc = c & 63;
    unsigned long long ga =
        (unsigned long long)(&H2[(size_t)(m0 + r) * HDIM + cc * 8]);
    unsigned la = (unsigned)(uintptr_t)(&As[r * HLD + cc * 8]);
    asm volatile("global_load_async_to_lds_b128 %0, %1, off"
                 :: "v"(la), "v"(ga) : "memory");
  }
  asm volatile("s_wait_asynccnt 0x0" ::: "memory");
  __syncthreads();
  const int w = tid >> 5, lane = tid & 31;
  const int n0 = blockIdx.x * 256 + w * 32;
  v8f acc[4][2];
  for (int mi = 0; mi < 4; ++mi) { acc[mi][0] = zero_v8f(); acc[mi][1] = zero_v8f(); }
  for (int kt = 0; kt < HDIM; kt += 32) {
    v16bf b0 = load_fragB(Wo + (size_t)n0 * HDIM, HDIM, kt);
    v16bf b1 = load_fragB(Wo + (size_t)(n0 + 16) * HDIM, HDIM, kt);
#pragma unroll
    for (int mi = 0; mi < 4; ++mi) {
      v16bf a = load_fragA(As + mi * 16 * HLD, HLD, kt);
      acc[mi][0] = wmma_bf16(a, b0, acc[mi][0]);
      acc[mi][1] = wmma_bf16(a, b1, acc[mi][1]);
    }
  }
  const int mrow = (lane >> 4) * 8, nc = lane & 15;
  for (int mi = 0; mi < 4; ++mi) {
    for (int r = 0; r < 8; ++r) {
      int g = m0 + mi * 16 + mrow + r;
      int t = g >> 3, b = g & 7;
      size_t base = ((size_t)b * SDEC + t) * (size_t)NVOC;
      int n = n0 + nc;
      out[base + n] = acc[mi][0][r] + bo[n];
      out[base + n + 16] = acc[mi][1][r] + bo[n + 16];
    }
  }
}

// --------------------------------- host ------------------------------------

extern "C" void kernel_launch(void* const* d_in, const int* in_sizes, int n_in,
                              void* d_out, int out_size, void* d_ws, size_t ws_size,
                              hipStream_t stream) {
  (void)in_sizes; (void)n_in; (void)out_size; (void)ws_size;
  const int*   enc_input = (const int*)  d_in[0];
  const int*   target    = (const int*)  d_in[1];
  const float* enc_emb   = (const float*)d_in[2];
  const float* enc_Wi1   = (const float*)d_in[3];
  const float* enc_bi1   = (const float*)d_in[4];
  const float* enc_Wh1   = (const float*)d_in[5];
  const float* enc_bh1   = (const float*)d_in[6];
  const float* enc_Wi2   = (const float*)d_in[7];
  const float* enc_bi2   = (const float*)d_in[8];
  const float* enc_Wh2   = (const float*)d_in[9];
  const float* enc_bh2   = (const float*)d_in[10];
  const float* dec_emb   = (const float*)d_in[11];
  const float* attn_We   = (const float*)d_in[12];
  const float* attn_be   = (const float*)d_in[13];
  const float* attn_Wh   = (const float*)d_in[14];
  const float* attn_bh   = (const float*)d_in[15];
  const float* attn_V    = (const float*)d_in[16];
  const float* attn_bV   = (const float*)d_in[17];
  const float* dec_Wi1   = (const float*)d_in[18];
  const float* dec_bi1   = (const float*)d_in[19];
  const float* dec_Wh1   = (const float*)d_in[20];
  const float* dec_bh1   = (const float*)d_in[21];
  const float* dec_Wi2   = (const float*)d_in[22];
  const float* dec_bi2   = (const float*)d_in[23];
  const float* dec_Wh2   = (const float*)d_in[24];
  const float* dec_bh2   = (const float*)d_in[25];
  const float* out_W     = (const float*)d_in[26];
  const float* out_b     = (const float*)d_in[27];
  float* out = (float*)d_out;

  char* p = (char*)d_ws;
  auto carve = [&](size_t bytes) -> char* {
    char* r = p;
    p += (bytes + 255) & ~(size_t)255;
    return r;
  };
  bf16_t* Wo_bf  = (bf16_t*)carve((size_t)NVOC * HDIM * 2);
  bf16_t* Xemb   = (bf16_t*)carve((size_t)BATCH * SENC * EDIM * 2);
  bf16_t* Wi1bf  = (bf16_t*)carve((size_t)HDIM * EDIM * 2);
  bf16_t* Wh1bf  = (bf16_t*)carve((size_t)HDIM * HDIM * 2);
  bf16_t* Wi2bf  = (bf16_t*)carve((size_t)HDIM * HDIM * 2);
  bf16_t* Wh2bf  = (bf16_t*)carve((size_t)HDIM * HDIM * 2);
  bf16_t* Webf   = (bf16_t*)carve((size_t)HDIM * HDIM * 2);
  bf16_t* Whabf  = (bf16_t*)carve((size_t)HDIM * HDIM * 2);
  bf16_t* Wcat1  = (bf16_t*)carve((size_t)HDIM * KCAT * 2);
  bf16_t* W2sum  = (bf16_t*)carve((size_t)HDIM * HDIM * 2);
  float*  bcat1  = (float*) carve((size_t)HDIM * 4);
  float*  b2sum  = (float*) carve((size_t)HDIM * 4);
  float*  xi1    = (float*) carve((size_t)BATCH * SENC * HDIM * 4);
  float*  xi2    = (float*) carve((size_t)BATCH * SENC * HDIM * 4);
  bf16_t* out1bf = (bf16_t*)carve((size_t)BATCH * SENC * HDIM * 2);
  bf16_t* encbf  = (bf16_t*)carve((size_t)BATCH * SENC * HDIM * 2);
  float*  preenc = (float*) carve((size_t)BATCH * SENC * HDIM * 4);
  float*  h1T    = (float*) carve((size_t)BATCH * HDIM * 4);
  float*  h2T    = (float*) carve((size_t)BATCH * HDIM * 4);
  bf16_t* H2     = (bf16_t*)carve((size_t)SDEC * BATCH * HDIM * 2);

  auto cvt = [&](const float* s, bf16_t* d, int n) {
    k_cvt_bf16<<<(n + 255) / 256, 256, 0, stream>>>(s, d, n);
  };
  cvt(out_W,   Wo_bf, NVOC * HDIM);
  cvt(enc_Wi1, Wi1bf, HDIM * EDIM);
  cvt(enc_Wh1, Wh1bf, HDIM * HDIM);
  cvt(enc_Wi2, Wi2bf, HDIM * HDIM);
  cvt(enc_Wh2, Wh2bf, HDIM * HDIM);
  cvt(attn_We, Webf,  HDIM * HDIM);
  cvt(attn_Wh, Whabf, HDIM * HDIM);
  k_prep_dec<<<(HDIM * KCAT + 255) / 256, 256, 0, stream>>>(
      dec_Wi1, dec_Wh1, dec_Wi2, dec_Wh2, dec_bi1, dec_bh1, dec_bi2, dec_bh2,
      Wcat1, W2sum, bcat1, b2sum);
  k_gather_emb<<<(BATCH * SENC * EDIM + 255) / 256, 256, 0, stream>>>(
      enc_input, enc_emb, Xemb, BATCH * SENC, EDIM);

  dim3 gH(HDIM / 256, BATCH * SENC / 16);
  // encoder layer 1: xi1 = emb @ Wi1^T + bi1 ; scan
  k_gemm_bias<<<gH, 256, 0, stream>>>(Xemb, Wi1bf, enc_bi1, xi1, HDIM, EDIM);
  k_rnn_scan<<<1, 256, 0, stream>>>(xi1, Wh1bf, enc_bh1, nullptr, out1bf, h1T, SENC);
  // encoder layer 2 (initial hidden = layer1 final): xi2 = out1 @ Wi2^T + bi2
  k_gemm_bias<<<gH, 256, 0, stream>>>(out1bf, Wi2bf, enc_bi2, xi2, HDIM, HDIM);
  k_rnn_scan<<<1, 256, 0, stream>>>(xi2, Wh2bf, enc_bh2, h1T, encbf, h2T, SENC);
  // pre_enc = enc_out @ attn_We^T + attn_be
  k_gemm_bias<<<gH, 256, 0, stream>>>(encbf, Webf, attn_be, preenc, HDIM, HDIM);

  size_t smemDec = (size_t)16 * A1LD * 2 + (size_t)16 * HLD * 2 +
                   (size_t)BATCH * HDIM * 2 + (size_t)BATCH * SENC * 4;
  k_decoder<<<1, 512, smemDec, stream>>>(target, dec_emb, preenc, encbf, Whabf,
                                         attn_bh, attn_V, attn_bV, Wcat1, bcat1,
                                         W2sum, b2sum, h2T, H2);

  size_t smemOut = (size_t)64 * HLD * 2;
  dim3 gO(NVOC / 256, SDEC * BATCH / 64);
  k_out_gemm<<<gO, 256, smemOut, stream>>>(H2, Wo_bf, out_b, out);
}
